// PLRNN_850403524609
// MI455X (gfx1250) — matrix operand order repaired and checked by reading, not compiled, not run on hardware
//
#include <hip/hip_runtime.h>
#include <hip/hip_bf16.h>

typedef __attribute__((ext_vector_type(16))) _Float16 v16h;
typedef __attribute__((ext_vector_type(8)))  _Float16 v8h;
typedef __attribute__((ext_vector_type(2)))  _Float16 h2v_t;   // packed half2
typedef __attribute__((ext_vector_type(8)))  float    v8f;
typedef __attribute__((ext_vector_type(4)))  float    v4f;

#define B_   512
#define T_   1024
#define DX   32
#define DZ   64
#define DY   256
#define TFA  0.125f

// Padded LDS row strides (in elements) to avoid bank conflicts on strided
// per-lane ds_load_b128 (stride mod 64 dwords != 0).
#define W2S  72    // halves per row of W2 tile [y=256][z=64]   (B operand GEMM1)
#define W1S  264   // halves per row of W1 tile [z=64][y=256]   (B operand GEMM2)
#define ZS   68    // floats per row of z state [m=16][z=64]
#define HS   264   // halves per row of H       [m=16][y=256]

// LDS byte layout
#define OFF_W2   0
#define SZ_W2    (256 * W2S * 2)          // 36864
#define OFF_W1   (OFF_W2 + SZ_W2)
#define SZ_W1    (64 * W1S * 2)           // 33792
#define OFF_Z    (OFF_W1 + SZ_W1)
#define SZ_Z     (16 * ZS * 4)            // 4352 per wave
#define OFF_H    (OFF_Z + 4 * SZ_Z)
#define SZ_H     (16 * HS * 2)            // 8448 per wave
#define LDS_TOTAL (OFF_H + 4 * SZ_H)      // 121856 bytes (< 320KB WGP limit)

union AFrag {
    v16h v;
    v8h  h[2];
};
union HFrag {
    v8h   v;
    h2v_t d[4];
};

// One workgroup of 4 wave32s per WGP; the serial recurrence gives only 32
// waves total, so occupancy is irrelevant -> let RA use the full VGPR budget.
__global__ void __launch_bounds__(128, 1)
PLRNN_850403524609_kernel(const float* __restrict__ X,
                          const float* __restrict__ A,
                          const float* __restrict__ W1,
                          const float* __restrict__ W2,
                          const float* __restrict__ h1,
                          const float* __restrict__ h2,
                          float* __restrict__ out)
{
    extern __shared__ char smem[];
    _Float16* w2h = (_Float16*)(smem + OFF_W2);   // [y][z] padded == B[n][k] GEMM1
    _Float16* w1h = (_Float16*)(smem + OFF_W1);   // [z][y] padded == B[n][k] GEMM2

    const int tid  = threadIdx.x;
    const int wid  = tid >> 5;        // wave in block (0..3)
    const int lane = tid & 31;
    const int ln   = lane & 15;       // n / m index within 16
    const int hl   = lane >> 4;       // lane-half selector

    float*    zF = (float*)   (smem + OFF_Z) + wid * (16 * ZS);
    _Float16* Hh = (_Float16*)(smem + OFF_H) + wid * (16 * HS);

    // ---- one-time: weights f32 -> f16 into LDS (whole block cooperates) ----
    for (int i = tid; i < 256 * 64; i += blockDim.x) {
        int y = i >> 6, z = i & 63;
        w2h[y * W2S + z] = (_Float16)W2[i];       // W2 is (dy=256, dz=64) row-major
    }
    for (int i = tid; i < 64 * 256; i += blockDim.x) {
        int z = i >> 8, y = i & 255;
        w1h[z * W1S + y] = (_Float16)W1[i];       // W1 is (dz=64, dy=256) row-major
    }

    // ---- per-lane broadcast constants: bias as ONE packed half2 dword per
    //      tile (v_pk_add_f16 reuses the single reg) ----
    h2v_t h2p[16];
#pragma unroll
    for (int j = 0; j < 16; ++j) {
        _Float16 hv = (_Float16)h2[16 * j + ln];
        h2v_t p; p[0] = hv; p[1] = hv;
        h2p[j] = p;
    }
    float h1t[4], At[4];
#pragma unroll
    for (int jj = 0; jj < 4; ++jj) { h1t[jj] = h1[16 * jj + ln]; At[jj] = A[16 * jj + ln]; }

    const int r0 = (blockIdx.x * 4 + wid) * 16;   // batch-row base of this wave

    // ---- loop-invariant per-lane base pointers ----
    float*          zfb  = zF + ln * ZS + 16 * hl;                          // z0 init
    const float*    zab  = zF + ln * ZS + 8 * hl;                           // GEMM1 A build
    const float*    zdb  = zF + 8 * hl * ZS + ln;                           // D-layout reads
    float*          zwb  = zF + 8 * hl * ZS + ln;                           // D-layout writes
    const _Float16* w2b  = w2h + ln * W2S + 8 * hl;                         // GEMM1 B
    const _Float16* w1b  = w1h + ln * W1S + 8 * hl;                         // GEMM2 B
    _Float16*       hwb  = Hh + 8 * hl * HS + ln;                           // H stores
    const _Float16* hab  = Hh + ln * HS + 8 * hl;                           // GEMM2 A
    float*          ob   = out + ((size_t)(r0 + 8 * hl) * T_) * DX + ln;    // output
    const float*    xdb  = X   + ((size_t)(r0 + 8 * hl) * T_) * DX + ln;    // x in D-layout

    // H-tile epilogue: bias (pk_add with one dword) + relu (pk_max vs inline 0)
    auto h_epi = [&](int jq, v8h cc) {
        HFrag u; u.v = cc;
#pragma unroll
        for (int i = 0; i < 4; ++i) u.d[i] += h2p[jq];
        v8h zero = {};
        v8h dd = __builtin_elementwise_max(u.v, zero);
#pragma unroll
        for (int r = 0; r < 8; ++r)
            hwb[r * HS + 16 * jq] = dd[r];
    };

    // ---- z0 = teacher_force(zeros, X[:,0], alpha=1); zf(step0) == z0 ----
    {
        const float* xp = X + ((size_t)(r0 + ln) * T_) * DX;
#pragma unroll
        for (int q = 0; q < 16; ++q) {
            float x = xp[16 * hl + q];
            zfb[q]      = (x != x) ? 0.0f : x;    // cols 0..31 (NaN -> 0)
            zfb[32 + q] = 0.0f;                   // cols 32..63
        }
    }

    // ---- x double-buffer (D-layout): holds x_{t+1}, refilled a step ahead ----
    float xb[2][8];
    {
        const float* xd = xdb + (size_t)1 * DX;   // x_1
#pragma unroll
        for (int jj = 0; jj < 2; ++jj)
#pragma unroll
            for (int r = 0; r < 8; ++r)
                xb[jj][r] = xd[(size_t)r * T_ * DX + 16 * jj];
    }
    __syncthreads();   // weights ready

    for (int t = 0; t < T_; ++t) {
        // ---- (b) GEMM1 A-fragments: zf (f32, LDS) -> f16 ISA A-layout ----
        AFrag a1[2];
#pragma unroll
        for (int kb = 0; kb < 2; ++kb) {
            v4f f0 = *(const v4f*)(zab + kb * 32);
            v4f f1 = *(const v4f*)(zab + kb * 32 + 4);
            v4f f2 = *(const v4f*)(zab + kb * 32 + 16);
            v4f f3 = *(const v4f*)(zab + kb * 32 + 20);
#pragma unroll
            for (int e = 0; e < 4; ++e) {
                a1[kb].h[0][e]     = (_Float16)f0[e];
                a1[kb].h[0][e + 4] = (_Float16)f1[e];
                a1[kb].h[1][e]     = (_Float16)f2[e];
                a1[kb].h[1][e + 4] = (_Float16)f3[e];
            }
        }

        // ---- (c) H = relu(Zf @ W2^T + h2); f16 accumulate, depth-2 pipeline:
        //      epilogue(j-2) runs before tile j's WMMAs -> hazard nops filled.
        v8h cq[2];
#pragma unroll
        for (int j = 0; j < 16; ++j) {
            if (j >= 2) h_epi(j - 2, cq[j & 1]);
            AFrag b0, b1;                    // both K-fragments loaded up front
            b0.h[0] = *(const v8h*)(w2b + j * (16 * W2S));
            b0.h[1] = *(const v8h*)(w2b + j * (16 * W2S) + 16);
            b1.h[0] = *(const v8h*)(w2b + j * (16 * W2S) + 32);
            b1.h[1] = *(const v8h*)(w2b + j * (16 * W2S) + 48);
            v8h c = {};   // -> v_wmma ... , 0 (inline SRC2)
            c = __builtin_amdgcn_wmma_f16_16x16x32_f16(false, a1[0].v, false, b0.v,
                                                       (short)0, c, false, false);
            c = __builtin_amdgcn_wmma_f16_16x16x32_f16(false, a1[1].v, false, b1.v,
                                                       (short)0, c, false, false);
            cq[j & 1] = c;
        }
        h_epi(14, cq[0]);
        h_epi(15, cq[1]);

        // ---- (d) Z' = A .* zf + H @ W1^T + h1  (f32 accumulate / f32 state) ----
        v8f acc[4];
#pragma unroll
        for (int jj = 0; jj < 4; ++jj) {
#pragma unroll
            for (int r = 0; r < 8; ++r)
                acc[jj][r] = fmaf(At[jj], zdb[r * ZS + 16 * jj], h1t[jj]);
        }
#pragma unroll
        for (int kb = 0; kb < 8; ++kb) {
            AFrag a2, b[4];                  // batch all loads, then 4 WMMAs
            a2.h[0] = *(const v8h*)(hab + kb * 32);
            a2.h[1] = *(const v8h*)(hab + kb * 32 + 16);
#pragma unroll
            for (int jj = 0; jj < 4; ++jj) {
                b[jj].h[0] = *(const v8h*)(w1b + jj * (16 * W1S) + kb * 32);
                b[jj].h[1] = *(const v8h*)(w1b + jj * (16 * W1S) + kb * 32 + 16);
            }
#pragma unroll
            for (int jj = 0; jj < 4; ++jj)   // 4 independent accumulation chains
                acc[jj] = __builtin_amdgcn_wmma_f32_16x16x32_f16(false, a2.v, false, b[jj].v,
                                                                 (short)0, acc[jj], false, false);
        }

        // ---- (e) fused epilogue: out = raw Z' ; zF = teacher_force(Z', x_{t+1})
        {
            float* op = ob + (size_t)t * DX;
#pragma unroll
            for (int jj = 0; jj < 4; ++jj) {
#pragma unroll
                for (int r = 0; r < 8; ++r) {
                    float zn = acc[jj][r];
                    if (jj < 2) {                 // n < 32: output + forcing
                        op[(size_t)r * T_ * DX + 16 * jj] = zn;
                        float xv = xb[jj][r];
                        float bl = fmaf(TFA, xv, (1.0f - TFA) * zn);
                        zn = (xv != xv) ? zn : bl;
                    }
                    zwb[r * ZS + 16 * jj] = zn;
                }
            }
        }
        // refill x buffer with x_{t+2} (consumed in next step's epilogue)
        if (t + 2 < T_) {
            const float* xd = xdb + (size_t)(t + 2) * DX;
#pragma unroll
            for (int jj = 0; jj < 2; ++jj)
#pragma unroll
                for (int r = 0; r < 8; ++r)
                    xb[jj][r] = xd[(size_t)r * T_ * DX + 16 * jj];
        }
    }
}

extern "C" void kernel_launch(void* const* d_in, const int* in_sizes, int n_in,
                              void* d_out, int out_size, void* d_ws, size_t ws_size,
                              hipStream_t stream) {
    (void)in_sizes; (void)n_in; (void)out_size; (void)d_ws; (void)ws_size;
    const float* X  = (const float*)d_in[0];
    const float* A  = (const float*)d_in[1];
    const float* W1 = (const float*)d_in[2];
    const float* W2 = (const float*)d_in[3];
    const float* h1 = (const float*)d_in[4];
    const float* h2 = (const float*)d_in[5];
    float* out = (float*)d_out;

    dim3 grid(B_ / 64);    // 8 blocks x 4 waves x 16 rows = 512 batch rows
    dim3 block(128);       // 4 wave32s
    hipLaunchKernelGGL(PLRNN_850403524609_kernel, grid, block, LDS_TOTAL, stream,
                       X, A, W1, W2, h1, h2, out);
}
